// Decoder_Light_67946382622907
// MI455X (gfx1250) — compile-verified, matrix operand order checked by reading, not run on hardware
//
#include <hip/hip_runtime.h>
#include <math.h>

typedef __attribute__((ext_vector_type(16))) _Float16 v16h;
typedef __attribute__((ext_vector_type(8)))  float    v8f;

#define BN 2
#define EPS 1e-5f

// ---------------------------------------------------------------------------
// Per-(b,c) instance-norm statistics: mean and rsqrt(var+eps)
// ---------------------------------------------------------------------------
__global__ __launch_bounds__(256)
void stats_kernel(const float* __restrict__ x, float* __restrict__ mean,
                  float* __restrict__ rstd, int HW)
{
    int bc = blockIdx.x;
    const float* p = x + (size_t)bc * HW;
    float s = 0.f, s2 = 0.f;
    for (int i = threadIdx.x; i < HW; i += 256) {
        float v = p[i];
        s += v; s2 += v * v;
    }
    __shared__ float sh[256], sh2[256];
    sh[threadIdx.x] = s; sh2[threadIdx.x] = s2;
    __syncthreads();
    for (int o = 128; o > 0; o >>= 1) {
        if (threadIdx.x < o) {
            sh[threadIdx.x]  += sh[threadIdx.x + o];
            sh2[threadIdx.x] += sh2[threadIdx.x + o];
        }
        __syncthreads();
    }
    if (threadIdx.x == 0) {
        float m = sh[0] / (float)HW;
        float v = sh2[0] / (float)HW - m * m;
        mean[bc] = m;
        rstd[bc] = rsqrtf(v + EPS);
    }
}

// ---------------------------------------------------------------------------
// out = inorm(in) (+ in if addres), written to f32 and/or f16
// ---------------------------------------------------------------------------
__global__ __launch_bounds__(256)
void norm_apply_kernel(const float* __restrict__ in, const float* __restrict__ mean,
                       const float* __restrict__ rstd, float* __restrict__ outf32,
                       _Float16* __restrict__ outf16, int HW, int addres, int total)
{
    int i = blockIdx.x * 256 + threadIdx.x;
    if (i >= total) return;
    int bc = i / HW;
    float v = in[i];
    float t = (v - mean[bc]) * rstd[bc];
    if (addres) t += v;
    if (outf32) outf32[i] = t;
    if (outf16) outf16[i] = (_Float16)t;
}

__global__ __launch_bounds__(256)
void cvt_f16_kernel(const float* __restrict__ in, _Float16* __restrict__ out, int total)
{
    int i = blockIdx.x * 256 + threadIdx.x;
    if (i < total) out[i] = (_Float16)in[i];
}

// ---------------------------------------------------------------------------
// Pack conv weights into f16 A-matrix [Mpad x K], K laid out k = r*Cin + ci
// (ci-major within each tap r, so a 32-wide WMMA K-tile has a single r).
// transposed=1: source is (Cin, Cout=Mreal, 3, 3)  (PAC wf = W.permute(1,0,2,3))
// ---------------------------------------------------------------------------
__global__ __launch_bounds__(256)
void pack_w_kernel(const float* __restrict__ W, _Float16* __restrict__ A,
                   int Mreal, int Mpad, int K, int Cin, int transposed)
{
    int i = blockIdx.x * 256 + threadIdx.x;
    if (i >= Mpad * K) return;
    int m = i / K, k = i - m * K;
    int r = k / Cin, ci = k - r * Cin;
    int ksz = K / Cin;
    float v = 0.f;
    if (m < Mreal) {
        if (!transposed) {
            v = W[((size_t)m * Cin + ci) * ksz + r];
        } else {
            v = W[((size_t)ci * Mreal + m) * 9 + r];
        }
    }
    A[i] = (_Float16)v;
}

// ---------------------------------------------------------------------------
// Gaussian pixel-adaptive kernel: out[b,r,h,w] = exp(-0.5 * sum_c (g_nbr - g_ctr)^2)
// ---------------------------------------------------------------------------
__global__ __launch_bounds__(256)
void pac_kernel(const float* __restrict__ g, _Float16* __restrict__ out,
                int Cg, int H, int W)
{
    int idx = blockIdx.x * 256 + threadIdx.x;
    int total = BN * H * W;
    if (idx >= total) return;
    int w = idx % W; int t = idx / W; int h = t % H; int b = t / H;
    float s[9];
#pragma unroll
    for (int r = 0; r < 9; r++) s[r] = 0.f;
    const float* gb = g + (size_t)b * Cg * H * W;
    for (int ci = 0; ci < Cg; ci++) {
        const float* gc = gb + (size_t)ci * H * W;
        float c = gc[h * W + w];
#pragma unroll
        for (int r = 0; r < 9; r++) {
            if (r == 4) continue;
            int y = h + r / 3 - 1, x = w + r % 3 - 1;
            float p = (y >= 0 && y < H && x >= 0 && x < W) ? gc[y * W + x] : 0.f;
            float d = p - c;
            s[r] += d * d;
        }
    }
#pragma unroll
    for (int r = 0; r < 9; r++) {
        float kv = (r == 4) ? 1.f : expf(-0.5f * s[r]);
        out[(((size_t)(b * 9 + r)) * H + h) * W + w] = (_Float16)kv;
    }
}

// ---------------------------------------------------------------------------
// Fragment K-offset, lane-half folded into base pointers:
//   koff(e,hf) = hf*8 + koffc(e),  koffc(e) = (e/8)*16 + (e%8)  (compile-time)
// ---------------------------------------------------------------------------
__device__ __forceinline__ constexpr int koffc_of(int e) {
    return ((e >> 3) << 4) + (e & 7);
}

// ---------------------------------------------------------------------------
// Implicit-GEMM conv (KSZ = 9: 3x3 pad 1, KSZ = 1: 1x1 pad 0), f16 in, f32 acc.
// K layout: k = r*CIN + ci, CIN % 32 == 0 -> each WMMA K-tile has a single tap.
// Compile-time CIN/H/W -> B-element loads use immediate offsets.
// One wave computes a 16x16 output tile.
// ---------------------------------------------------------------------------
template <int KSZ, int CIN, int H, int W>
__global__ __launch_bounds__(256)
void conv_wmma_kernel(const _Float16* __restrict__ in16,
                      const _Float16* __restrict__ Apack,
                      const float* __restrict__ bias,
                      const float* __restrict__ res,
                      float* __restrict__ outf32,
                      _Float16* __restrict__ outf16,
                      int Mpad, int Mreal)
{
    constexpr int K = KSZ * CIN;
    constexpr int HWs = H * W;
    constexpr int N = BN * HWs;
    constexpr int tiles_n = N >> 4;
    const int tiles_m = Mpad >> 4;
    int wid = (blockIdx.x * 256 + threadIdx.x) >> 5;
    if (wid >= tiles_m * tiles_n) return;
    int tm = wid / tiles_n, tn = wid - tm * tiles_n;
    int lane = threadIdx.x & 31;
    int hf = lane >> 4, ln = lane & 15;

    int n = tn * 16 + ln;
    int w = n % W; int t = n / W; int h = t % H; int b = t / H;

    const _Float16* inb  = in16 + (size_t)b * CIN * HWs + hf * 8 * HWs;
    const _Float16* Arow = Apack + (size_t)(tm * 16 + ln) * K + hf * 8;
    v8f acc = {};

#pragma unroll
    for (int r = 0; r < KSZ; r++) {
        bool valid;
        int y, x;
        if (KSZ == 9) {
            y = h + r / 3 - 1; x = w + r % 3 - 1;
            valid = (y >= 0 && y < H && x >= 0 && x < W);
        } else {
            y = h; x = w; valid = true;
        }
        int ys = valid ? y : 0, xs = valid ? x : 0;
        const _Float16* pb = inb + ys * W + xs;
        const _Float16* ap = Arow + r * CIN;
#pragma unroll 1
        for (int c0 = 0; c0 < CIN; c0 += 32, pb += 32 * HWs, ap += 32) {
            __builtin_prefetch(ap + 64, 0, 1);
            v16h a, bm;
#pragma unroll
            for (int e = 0; e < 16; e++) a[e] = ap[koffc_of(e)];
#pragma unroll
            for (int e = 0; e < 16; e++)
                bm[e] = valid ? pb[koffc_of(e) * HWs] : (_Float16)0.f;
            acc = __builtin_amdgcn_wmma_f32_16x16x32_f16(false, a, false, bm,
                                                         (short)0, acc, false, false);
        }
    }
#pragma unroll
    for (int r = 0; r < 8; r++) {
        int m = tm * 16 + hf * 8 + r;
        if (m < Mreal) {
            size_t oi = ((size_t)(b * Mreal + m) * H + h) * W + w;
            float v = acc[r] + bias[m];
            if (res) v += res[oi];
            if (outf32) outf32[oi] = v;
            if (outf16) outf16[oi] = (_Float16)v;
        }
    }
}

// ---------------------------------------------------------------------------
// PacConvTranspose2d (k=3, stride=2, pad=1, out_pad=1) as implicit GEMM.
// B-element = zero-upsampled input * per-pixel Gaussian kernel (f16 mul).
// Taps with even y = h+kh are wave-uniformly zero and skipped entirely.
// ---------------------------------------------------------------------------
template <int CIN, int HIN, int WIN>
__global__ __launch_bounds__(256)
void pacconvT_wmma_kernel(const _Float16* __restrict__ in16,   // (B,CIN,HIN,WIN)
                          const _Float16* __restrict__ pk16,   // (B,9,2H,2W)
                          const _Float16* __restrict__ Apack,  // (M x K)
                          const float* __restrict__ bias,
                          float* __restrict__ outf32,
                          int M)
{
    constexpr int K = CIN * 9;
    constexpr int Hout = HIN * 2, Wout = WIN * 2;
    constexpr int HWi = HIN * WIN;
    constexpr int HWo = Hout * Wout;
    constexpr int N = BN * HWo;
    constexpr int tiles_n = N >> 4;
    const int tiles_m = M >> 4;
    int wid = (blockIdx.x * 256 + threadIdx.x) >> 5;
    if (wid >= tiles_m * tiles_n) return;
    int tm = wid / tiles_n, tn = wid - tm * tiles_n;
    int lane = threadIdx.x & 31;
    int hf = lane >> 4, ln = lane & 15;

    int n = tn * 16 + ln;
    int w = n % Wout; int t = n / Wout; int h = t % Hout; int b = t / Hout;

    const _Float16* inb  = in16 + (size_t)b * CIN * HWi + hf * 8 * HWi;
    const _Float16* pkb  = pk16 + (size_t)b * 9 * HWo + (size_t)h * Wout + w;
    const _Float16* Arow = Apack + (size_t)(tm * 16 + ln) * K + hf * 8;
    v8f acc = {};

#pragma unroll
    for (int r = 0; r < 9; r++) {
        constexpr int khs[9] = {0,0,0,1,1,1,2,2,2};
        const int kh = khs[r], kw = r - kh * 3;
        int y = h + kh;                       // into padded upsampled grid
        if ((y & 1) == 0) continue;           // wave-uniform: whole B tile zero
        int x = w + kw;
        int yy = (y - 1) >> 1, xx = (x - 1) >> 1;
        bool valid = (x & 1) && (yy < HIN) && (xx < WIN);
        int ys = valid ? yy : 0, xs = valid ? xx : 0;
        // Gaussian kernel: address always in-bounds -> unconditional load
        _Float16 pkv = pkb[r * HWo];
        _Float16 sch = valid ? pkv : (_Float16)0.f;
        const _Float16* pb = inb + ys * WIN + xs;
        const _Float16* ap = Arow + r * CIN;
#pragma unroll 1
        for (int c0 = 0; c0 < CIN; c0 += 32, pb += 32 * HWi, ap += 32) {
            __builtin_prefetch(ap + 64, 0, 1);
            v16h a, bm;
#pragma unroll
            for (int e = 0; e < 16; e++) a[e] = ap[koffc_of(e)];
#pragma unroll
            for (int e = 0; e < 16; e++)
                bm[e] = pb[koffc_of(e) * HWi] * sch;
            acc = __builtin_amdgcn_wmma_f32_16x16x32_f16(false, a, false, bm,
                                                         (short)0, acc, false, false);
        }
    }
#pragma unroll
    for (int r = 0; r < 8; r++) {
        int m = tm * 16 + hf * 8 + r;
        size_t oi = ((size_t)(b * M + m) * Hout + h) * Wout + w;
        outf32[oi] = acc[r] + bias[m];
    }
}

// ---------------------------------------------------------------------------
extern "C" void kernel_launch(void* const* d_in, const int* in_sizes, int n_in,
                              void* d_out, int out_size, void* d_ws, size_t ws_size,
                              hipStream_t stream)
{
    const float* x      = (const float*)d_in[0];   // (2,256,64,64)
    const float* ef2    = (const float*)d_in[1];   // (2,64,128,128)
    const float* ef1    = (const float*)d_in[2];   // (2,32,256,256)
    const float* W_down = (const float*)d_in[3];
    const float* b_down = (const float*)d_in[4];
    const float* W_mid  = (const float*)d_in[5];
    const float* b_mid  = (const float*)d_in[6];
    const float* W_adj2 = (const float*)d_in[7];
    const float* b_adj2 = (const float*)d_in[8];
    const float* W_adj1 = (const float*)d_in[9];
    const float* b_adj1 = (const float*)d_in[10];
    const float* W16    = (const float*)d_in[11];
    const float* b16    = (const float*)d_in[12];
    const float* W20    = (const float*)d_in[13];
    const float* b20    = (const float*)d_in[14];
    const float* W24    = (const float*)d_in[15];
    const float* b24    = (const float*)d_in[16];
    float* out = (float*)d_out;

    // bump allocator over d_ws
    size_t off = 0;
    auto alloc = [&](size_t bytes) -> void* {
        void* p = (char*)d_ws + off;
        off += (bytes + 255) & ~(size_t)255;
        return p;
    };
    float*    mean    = (float*)alloc(512 * 4);
    float*    rstd    = (float*)alloc(512 * 4);
    _Float16* xn16    = (_Float16*)alloc((size_t)2*256*64*64 * 2);
    _Float16* ef2_16  = (_Float16*)alloc((size_t)2*64*128*128 * 2);
    _Float16* ef1_16  = (_Float16*)alloc((size_t)2*32*256*256 * 2);
    _Float16* A_down  = (_Float16*)alloc((size_t)128*2304 * 2);
    _Float16* A_mid   = (_Float16*)alloc((size_t)128*128 * 2);
    _Float16* A_adj2  = (_Float16*)alloc((size_t)128*576 * 2);
    _Float16* A_16    = (_Float16*)alloc((size_t)64*1152 * 2);
    _Float16* A_adj1  = (_Float16*)alloc((size_t)64*288 * 2);
    _Float16* A_20    = (_Float16*)alloc((size_t)32*576 * 2);
    _Float16* A_24    = (_Float16*)alloc((size_t)16*288 * 2);
    float*    t1      = (float*)alloc((size_t)2*128*64*64 * 4);
    _Float16* t1_16   = (_Float16*)alloc((size_t)2*128*64*64 * 2);
    _Float16* t2_16   = (_Float16*)alloc((size_t)2*128*64*64 * 2);
    float*    g2      = (float*)alloc((size_t)2*128*128*128 * 4);
    _Float16* pac2_16 = (_Float16*)alloc((size_t)2*9*128*128 * 2);
    float*    u16     = (float*)alloc((size_t)2*64*128*128 * 4);
    _Float16* u16_16  = (_Float16*)alloc((size_t)2*64*128*128 * 2);
    float*    g1      = (float*)alloc((size_t)2*64*256*256 * 4);
    _Float16* pac1_16 = (_Float16*)alloc((size_t)2*9*256*256 * 2);
    float*    u20     = (float*)alloc((size_t)2*32*256*256 * 4);
    _Float16* u20_16  = (_Float16*)alloc((size_t)2*32*256*256 * 2);

    auto cdiv = [](int a, int b) { return (a + b - 1) / b; };

    // ---- pack all weights to f16 A-matrices (k = r*Cin + ci layout) -------
    pack_w_kernel<<<cdiv(128*2304,256),256,0,stream>>>(W_down, A_down, 128,128,2304,256, 0);
    pack_w_kernel<<<cdiv(128*128,256), 256,0,stream>>>(W_mid,  A_mid,  128,128,128, 128, 0);
    pack_w_kernel<<<cdiv(128*576,256), 256,0,stream>>>(W_adj2, A_adj2, 128,128,576,  64, 0);
    pack_w_kernel<<<cdiv(64*1152,256), 256,0,stream>>>(W16,    A_16,   64, 64, 1152,128, 1);
    pack_w_kernel<<<cdiv(64*288,256),  256,0,stream>>>(W_adj1, A_adj1, 64, 64, 288,  32, 0);
    pack_w_kernel<<<cdiv(32*576,256),  256,0,stream>>>(W20,    A_20,   32, 32, 576,  64, 1);
    pack_w_kernel<<<cdiv(16*288,256),  256,0,stream>>>(W24,    A_24,   3,  16, 288,  32, 0);

    // ---- guidance features to f16 -----------------------------------------
    cvt_f16_kernel<<<cdiv(2*64*128*128,256),256,0,stream>>>(ef2, ef2_16, 2*64*128*128);
    cvt_f16_kernel<<<cdiv(2*32*256*256,256),256,0,stream>>>(ef1, ef1_16, 2*32*256*256);

    // ---- layer13: inorm(x) -> conv 3x3 256->128 ---------------------------
    stats_kernel<<<512,256,0,stream>>>(x, mean, rstd, 64*64);
    norm_apply_kernel<<<cdiv(2*256*64*64,256),256,0,stream>>>(x, mean, rstd,
                       nullptr, xn16, 64*64, 0, 2*256*64*64);
    conv_wmma_kernel<9,256,64,64><<<512,256,0,stream>>>(xn16, A_down, b_down,
                       nullptr, t1, t1_16, 128, 128);

    // ---- middle 1x1 + residual --------------------------------------------
    conv_wmma_kernel<1,128,64,64><<<512,256,0,stream>>>(t1_16, A_mid, b_mid,
                       t1, nullptr, t2_16, 128, 128);

    // ---- adjust_ef_lv2 then PAC kernel at 128x128 -------------------------
    conv_wmma_kernel<9,64,128,128><<<2048,256,0,stream>>>(ef2_16, A_adj2, b_adj2,
                       nullptr, g2, nullptr, 128, 128);
    pac_kernel<<<cdiv(2*128*128,256),256,0,stream>>>(g2, pac2_16, 128, 128, 128);

    // ---- layer16: PacConvTranspose 128->64, x2 ----------------------------
    pacconvT_wmma_kernel<128,64,64><<<1024,256,0,stream>>>(t2_16, pac2_16, A_16, b16,
                       u16, 64);

    // ---- layer17: inorm+res twice -----------------------------------------
    stats_kernel<<<128,256,0,stream>>>(u16, mean, rstd, 128*128);
    norm_apply_kernel<<<cdiv(2*64*128*128,256),256,0,stream>>>(u16, mean, rstd,
                       u16, nullptr, 128*128, 1, 2*64*128*128);
    stats_kernel<<<128,256,0,stream>>>(u16, mean, rstd, 128*128);
    norm_apply_kernel<<<cdiv(2*64*128*128,256),256,0,stream>>>(u16, mean, rstd,
                       u16, u16_16, 128*128, 1, 2*64*128*128);

    // ---- adjust_ef_lv1 then PAC kernel at 256x256 -------------------------
    conv_wmma_kernel<9,32,256,256><<<4096,256,0,stream>>>(ef1_16, A_adj1, b_adj1,
                       nullptr, g1, nullptr, 64, 64);
    pac_kernel<<<cdiv(2*256*256,256),256,0,stream>>>(g1, pac1_16, 64, 256, 256);

    // ---- layer20: PacConvTranspose 64->32, x2 -----------------------------
    pacconvT_wmma_kernel<64,128,128><<<2048,256,0,stream>>>(u16_16, pac1_16, A_20, b20,
                       u20, 32);

    // ---- layer21: inorm+res twice -----------------------------------------
    stats_kernel<<<64,256,0,stream>>>(u20, mean, rstd, 256*256);
    norm_apply_kernel<<<cdiv(2*32*256*256,256),256,0,stream>>>(u20, mean, rstd,
                       u20, nullptr, 256*256, 1, 2*32*256*256);
    stats_kernel<<<64,256,0,stream>>>(u20, mean, rstd, 256*256);
    norm_apply_kernel<<<cdiv(2*32*256*256,256),256,0,stream>>>(u20, mean, rstd,
                       u20, u20_16, 256*256, 1, 2*32*256*256);

    // ---- layer24: conv 3x3 32->3 ------------------------------------------
    conv_wmma_kernel<9,32,256,256><<<1024,256,0,stream>>>(u20_16, A_24, b24,
                       nullptr, out, nullptr, 16, 3);

    (void)in_sizes; (void)n_in; (void)out_size; (void)ws_size;
}
